// EpisodicMemory_32263794327943
// MI455X (gfx1250) — matrix operand order repaired and checked by reading, not compiled, not run on hardware
//
#include <hip/hip_runtime.h>
#include <hip/hip_bf16.h>
#include <stdint.h>

// ---------------------------------------------------------------------------
// Episodic memory retrieval for MI455X (gfx1250, wave32).
//   sims = (q/||q||) @ (k/||k||)^T  * (success>=0.7 | !any)   [256 x 500000]
//   top-32 per query  -> gather keys [B,K,128], values [B,K,256], success [B,K]
//
// Design: keys (256 MB) are streamed exactly once. One 256-thread workgroup
// holds ALL 256 normalized bf16 queries in VGPRs (A fragments), stages 32-key
// tiles in LDS (normalize+mask+cvt to bf16), and runs v_wmma_f32_16x16x32_bf16.
// Top-32 is fused via a wave-level bitonic K-selection with a ballot-gated
// threshold test. Chunk partials are merged, then gathered with float4 copies.
// ---------------------------------------------------------------------------

typedef __attribute__((ext_vector_type(16))) __bf16       v16bf;
typedef __attribute__((ext_vector_type(8)))  float        v8f;
typedef __attribute__((ext_vector_type(4)))  unsigned int u32x4;
typedef __attribute__((ext_vector_type(4)))  float        f32x4;

#define EPSN     1e-8f
#define THRESH   0.7f
#define NQ       256      // queries (B)
#define DIM      128      // key dim
#define VDIM     256      // value dim (2*DIM)
#define KTOP     32       // top_k
#define KB_CHUNK 4096     // keys per workgroup chunk

union FragBF { v16bf v; u32x4 u[2]; };

__device__ __forceinline__ unsigned short f2bf(float f) {
  union { float f; unsigned int u; } c; c.f = f;
  unsigned int u = c.u;
  u += 0x7FFFu + ((u >> 16) & 1u);          // round-to-nearest-even
  return (unsigned short)(u >> 16);
}
__device__ __forceinline__ unsigned int pack2(float lo, float hi) {
  return (unsigned int)f2bf(lo) | ((unsigned int)f2bf(hi) << 16);
}
__device__ __forceinline__ bool any32(bool p) {
  return __builtin_amdgcn_ballot_w32(p) != 0u;
}
__device__ __forceinline__ float sq4(f32x4 v) {
  return v.x * v.x + v.y * v.y + v.z * v.z + v.w * v.w;
}

// compare-exchange across lanes with index payload
__device__ __forceinline__ void ce(float& v, int& i, int j, bool wantLarger) {
  float ov = __shfl_xor(v, j, 32);
  int   oi = __shfl_xor(i, j, 32);
  bool take = wantLarger ? (ov > v) : (ov < v);
  if (take) { v = ov; i = oi; }
}

// (lv,li): per-query top-32 list, sorted ASCENDING across the 32 lanes.
// (cv,ci): one candidate per lane. Ballot-gated bitonic merge.
__device__ __forceinline__ void topk_update(float& lv, int& li, float cv, int ci, int lane) {
  float thr = __shfl(lv, 0, 32);            // current 32nd-best (list min)
  if (!any32(cv > thr)) return;             // wave-uniform skip
  // bitonic sort candidates DESCENDING
  #pragma unroll 1
  for (int k = 2; k <= 32; k <<= 1) {
    #pragma unroll 1
    for (int j = k >> 1; j > 0; j >>= 1) {
      bool low  = ((lane & j) == 0);
      bool desc = ((lane & k) == 0);
      ce(cv, ci, j, desc == low);
    }
  }
  // elementwise max(list_asc, batch_desc) -> bitonic sequence holding top-32 of union
  if (cv > lv) { lv = cv; li = ci; }
  // bitonic merge back to ASCENDING
  #pragma unroll 1
  for (int j = 16; j > 0; j >>= 1) {
    bool low = ((lane & j) == 0);
    ce(lv, li, j, !low);
  }
}

// ---------------------------------------------------------------------------
// Kernel 0a: any(success >= 0.7) flag (benign same-value race; memset'd to 0).
__global__ void valid_any_kernel(const float* __restrict__ sr,
                                 unsigned int* __restrict__ flag, int N) {
  bool v = false;
  for (int i = blockIdx.x * blockDim.x + threadIdx.x; i < N;
       i += gridDim.x * blockDim.x)
    v |= (sr[i] >= THRESH);
  if (v) flag[0] = 1u;
}

// Kernel 0b: normalize queries, convert to bf16 (row-major ushort[256][128]).
__global__ __launch_bounds__(128) void qnorm_kernel(const float* __restrict__ q,
                                                    unsigned short* __restrict__ qn) {
  __shared__ float lsum[4];
  int b = blockIdx.x, t = threadIdx.x;
  int lane = t & 31, wv = t >> 5;
  float x = q[b * DIM + t];
  float p = x * x;
  #pragma unroll
  for (int j = 16; j > 0; j >>= 1) p += __shfl_xor(p, j, 32);
  if (lane == 0) lsum[wv] = p;
  __syncthreads();
  float s  = lsum[0] + lsum[1] + lsum[2] + lsum[3];
  float rn = 1.0f / fmaxf(sqrtf(s), EPSN);
  qn[b * DIM + t] = f2bf(x * rn);
}

// ---------------------------------------------------------------------------
// Kernel 1: fused normalize+mask GEMM (bf16 WMMA) + running top-32 per query.
// grid.x = #chunks; 256 threads = 8 waves; wave w owns queries [32w, 32w+32).
__global__ __launch_bounds__(256) void sim_topk_kernel(
    const float* __restrict__ keys, const float* __restrict__ sr,
    const unsigned short* __restrict__ qn, const unsigned int* __restrict__ flag,
    float* __restrict__ pval, int* __restrict__ pidx, int N) {
  __shared__ unsigned short kb[32][136];   // 32 keys x 128 dims bf16, padded row pitch
  __shared__ float sq[32][8];
  __shared__ float scl[32];

  const int tid  = threadIdx.x;
  const int lane = tid & 31;
  const int wv   = tid >> 5;
  const int hi   = lane >> 4;              // half-wave select
  const int m16  = lane & 15;
  const int chunk = blockIdx.x;
  const int cbase = chunk * KB_CHUNK;
  const bool allValid = (*flag == 0u);

  // A fragments: 16-bit A 16x32 layout. lanes 0-15 row=lane, K {e<8: e, else e+8};
  // lanes 16-31 same rows, K offset +8. Two 16B loads per fragment.
  FragBF A[2][4];
  #pragma unroll
  for (int mt = 0; mt < 2; ++mt) {
    #pragma unroll
    for (int kf = 0; kf < 4; ++kf) {
      int row = wv * 32 + mt * 16 + m16;
      int off = kf * 32 + hi * 8;
      const u32x4* p = (const u32x4*)(qn + row * DIM + off);
      A[mt][kf].u[0] = p[0];               // K off..off+7
      A[mt][kf].u[1] = p[2];               // K off+16..off+23
    }
  }

  float lv[32]; int li[32];
  #pragma unroll
  for (int q = 0; q < 32; ++q) { lv[q] = -__builtin_inff(); li[q] = 0; }

  int rem   = N - cbase;                   // > 0 by grid construction
  int iters = (rem + 31) >> 5;
  if (iters > KB_CHUNK / 32) iters = KB_CHUNK / 32;

  for (int it = 0; it < iters; ++it) {
    const int kb0 = cbase + it * 32;

    // ---- stage: 32 keys -> normalize, success-mask, cvt bf16 -> LDS ----
    __syncthreads();                       // protect previous tile's consumers
    const int key = tid >> 3;              // 8 threads per key
    const int seg = tid & 7;               // 16 dims per thread
    const int grow = kb0 + key;
    f32x4 x0 = {}, x1 = {}, x2 = {}, x3 = {};
    if (grow < N) {
      const f32x4* kp = (const f32x4*)(keys + grow * DIM + seg * 16);
      x0 = kp[0]; x1 = kp[1]; x2 = kp[2]; x3 = kp[3];
    }
    sq[key][seg] = sq4(x0) + sq4(x1) + sq4(x2) + sq4(x3);
    __syncthreads();
    if (tid < 32) {
      float t = 0.f;
      #pragma unroll
      for (int i = 0; i < 8; ++i) t += sq[tid][i];
      float rn = 1.0f / fmaxf(sqrtf(t), EPSN);
      int g = kb0 + tid;
      float v = (g < N) ? sr[g] : 0.0f;
      scl[tid] = (allValid || v >= THRESH) ? rn : 0.0f;  // mask folded into scale
    }
    __syncthreads();
    {
      float c = scl[key];
      u32x4 o0, o1;
      o0.x = pack2(x0.x * c, x0.y * c); o0.y = pack2(x0.z * c, x0.w * c);
      o0.z = pack2(x1.x * c, x1.y * c); o0.w = pack2(x1.z * c, x1.w * c);
      o1.x = pack2(x2.x * c, x2.y * c); o1.y = pack2(x2.z * c, x2.w * c);
      o1.z = pack2(x3.x * c, x3.y * c); o1.w = pack2(x3.z * c, x3.w * c);
      u32x4* dst = (u32x4*)&kb[key][seg * 16];
      dst[0] = o0; dst[1] = o1;
    }
    __syncthreads();

    // ---- WMMA: this wave's 32 queries x 32 keys, K=128 ----
    // B 32x16 layout: lane col n=lane%16; same K order as A (lanes 0-15 K {0..7,16..23}, lanes 16-31 K +8).
    v8f C[2][2];
    #pragma unroll
    for (int nt = 0; nt < 2; ++nt) {
      FragBF Bf[4];
      int krow = nt * 16 + m16;
      #pragma unroll
      for (int kf = 0; kf < 4; ++kf) {
        const u32x4* bp = (const u32x4*)&kb[krow][kf * 32 + hi * 8];
        Bf[kf].u[0] = bp[0];
        Bf[kf].u[1] = bp[2];
      }
      #pragma unroll
      for (int mt = 0; mt < 2; ++mt) {
        v8f acc = {};
        #pragma unroll
        for (int kf = 0; kf < 4; ++kf)
          acc = __builtin_amdgcn_wmma_f32_16x16x32_bf16(
              false, A[mt][kf].v, false, Bf[kf].v, (short)0, acc, false, false);
        C[nt][mt] = acc;
      }
    }

    // ---- fused running top-32 (lane L <-> key kb0+L) ----
    const int gk = kb0 + lane;
    #pragma unroll
    for (int q = 0; q < 32; ++q) {
      const int mt  = q >> 4;
      const int m   = q & 15;
      const int reg = m & 7;               // C: M=reg (lanes 0-15) / reg+8 (16-31)
      const int src = m16 + ((m & 8) ? 16 : 0);
      float a = __shfl(C[0][mt][reg], src, 32);
      float b = __shfl(C[1][mt][reg], src, 32);
      float cv = (lane < 16) ? a : b;
      if (gk >= N) cv = -__builtin_inff();
      topk_update(lv[q], li[q], cv, gk, lane);
    }
  }

  // ---- per-chunk partial top-32 -> workspace ----
  #pragma unroll
  for (int q = 0; q < 32; ++q) {
    int qq = wv * 32 + q;
    int o  = (chunk * NQ + qq) * KTOP + lane;
    pval[o] = lv[q];
    pidx[o] = li[q];
  }
}

// ---------------------------------------------------------------------------
// Kernel 2: merge chunk partials per query; emit final indices (descending) and
// the gathered success-rate output directly.
__global__ __launch_bounds__(32) void merge_kernel(
    const float* __restrict__ pval, const int* __restrict__ pidx,
    const float* __restrict__ sr, int* __restrict__ fidx,
    float* __restrict__ outS, int nchunk) {
  int q = blockIdx.x, lane = threadIdx.x;
  float lv = -__builtin_inff(); int li = 0;
  for (int c = 0; c < nchunk; ++c) {
    int o = (c * NQ + q) * KTOP + lane;
    topk_update(lv, li, pval[o], pidx[o], lane);
  }
  int fi = __shfl(li, 31 - lane, 32);      // rank r = lane (descending order)
  fidx[q * KTOP + lane] = fi;
  outS[q * KTOP + lane] = sr[fi];
}

// Kernel 3: gather keys [B,K,128] and values [B,K,256] with float4 copies.
__global__ __launch_bounds__(256) void gather_kernel(
    const float* __restrict__ keys, const float* __restrict__ vals,
    const int* __restrict__ fidx, float* __restrict__ outK,
    float* __restrict__ outV) {
  __shared__ int idx[KTOP];
  int q = blockIdx.x, t = threadIdx.x;
  if (t < KTOP) idx[t] = fidx[q * KTOP + t];
  __syncthreads();
  for (int i = t; i < KTOP * (DIM / 4); i += 256) {          // 1024 f32x4
    int r = i / (DIM / 4), c = i % (DIM / 4);
    ((f32x4*)(outK + (q * KTOP + r) * DIM))[c] =
        ((const f32x4*)(keys + idx[r] * DIM))[c];
  }
  for (int i = t; i < KTOP * (VDIM / 4); i += 256) {         // 2048 f32x4
    int r = i / (VDIM / 4), c = i % (VDIM / 4);
    ((f32x4*)(outV + (q * KTOP + r) * VDIM))[c] =
        ((const f32x4*)(vals + idx[r] * VDIM))[c];
  }
}

// ---------------------------------------------------------------------------
extern "C" void kernel_launch(void* const* d_in, const int* in_sizes, int n_in,
                              void* d_out, int out_size, void* d_ws, size_t ws_size,
                              hipStream_t stream) {
  const float* query = (const float*)d_in[0];
  const float* keys  = (const float*)d_in[1];
  const float* vals  = (const float*)d_in[2];
  const float* sr    = (const float*)d_in[3];
  const int N = in_sizes[3];                     // 500000 episodes
  const int nchunk = (N + KB_CHUNK - 1) / KB_CHUNK;

  // workspace layout (~8.3 MB): flag | qn bf16 | pval | pidx | fidx
  char* ws = (char*)d_ws;
  unsigned int*   flag = (unsigned int*)ws;
  unsigned short* qn   = (unsigned short*)(ws + 256);
  size_t pbytes = (size_t)nchunk * NQ * KTOP * 4;
  float* pval = (float*)(ws + 256 + (size_t)NQ * DIM * 2);
  int*   pidx = (int*)((char*)pval + pbytes);
  int*   fidx = (int*)((char*)pidx + pbytes);

  float* outK = (float*)d_out;
  float* outV = outK + (size_t)NQ * KTOP * DIM;
  float* outS = outV + (size_t)NQ * KTOP * VDIM;

  hipMemsetAsync(flag, 0, sizeof(unsigned int), stream);
  valid_any_kernel<<<256, 256, 0, stream>>>(sr, flag, N);
  qnorm_kernel<<<NQ, DIM, 0, stream>>>(query, qn);
  sim_topk_kernel<<<nchunk, 256, 0, stream>>>(keys, sr, qn, flag, pval, pidx, N);
  merge_kernel<<<NQ, KTOP, 0, stream>>>(pval, pidx, sr, fidx, outS, nchunk);
  gather_kernel<<<NQ, 256, 0, stream>>>(keys, vals, fidx, outK, outV);
}
